// HardSuperClassOutputLayers_33389075759669
// MI455X (gfx1250) — matrix operand order repaired and checked by reading, compiled, not run on hardware
//
#include <hip/hip_runtime.h>

// ---- problem constants -----------------------------------------------------
#define NROW   32768
#define DDIM   1024
#define SNUM   27
#define GNUM   3
#define CNUM   80
#define JTOT   428          // 27 + 81 + 320 real output columns from the GEMM
#define JPAD   448          // padded to multiple of 2*16*NI
#define OSTR   509          // 27 + 81 + 81 + 320 output row stride

#define BM 128
#define BN 448
#define BK 32
#define LDP (BK + 8)        // padded LDS row stride (40 ushorts = 80B): bank-conflict-free
#define NI 14               // WMMA column tiles per wave (BN / 2 / 16)

typedef __attribute__((ext_vector_type(16))) __bf16        v16bf;
typedef __attribute__((ext_vector_type(8)))  float         v8f;
typedef __attribute__((ext_vector_type(4)))  float         f32x4;
typedef __attribute__((ext_vector_type(4)))  unsigned int  u32x4;
typedef __attribute__((ext_vector_type(2)))  unsigned int  u32x2;

__device__ __forceinline__ unsigned short f2bf(float f) {
    unsigned int u = __float_as_uint(f);
    u += 0x7FFFu + ((u >> 16) & 1u);            // round to nearest even
    return (unsigned short)(u >> 16);
}

// ---- 1) pack W_cls / W_fg / W_bbox into bf16 [JPAD][DDIM] col-major + bias --
__global__ __launch_bounds__(256)
void pack_weights(const float* __restrict__ Wc, const float* __restrict__ bc,
                  const float* __restrict__ Wf, const float* __restrict__ bfg,
                  const float* __restrict__ Wb, const float* __restrict__ bb,
                  unsigned short* __restrict__ Wp, float* __restrict__ bp) {
    int idx = blockIdx.x * 256 + threadIdx.x;   // 0 .. JPAD*DDIM-1
    int j = idx >> 10;
    int k = idx & (DDIM - 1);
    float v = 0.0f;
    if (j < SNUM) {
        v = Wc[k * SNUM + j];
    } else if (j < SNUM + SNUM * GNUM) {
        int t = j - SNUM, s = t / GNUM, g = t % GNUM;   // W_fg[s][k][g]
        v = Wf[(s * DDIM + k) * GNUM + g];
    } else if (j < JTOT) {
        v = Wb[k * (CNUM * 4) + (j - (SNUM + SNUM * GNUM))];
    }
    Wp[j * DDIM + k] = f2bf(v);
    if (k == 0) {
        float b = 0.0f;
        if (j < SNUM)                  b = bc[j];
        else if (j < SNUM + SNUM*GNUM) b = bfg[j - SNUM];
        else if (j < JTOT)             b = bb[j - (SNUM + SNUM * GNUM)];
        bp[j] = b;
    }
}

// ---- 2) fused GEMM: [32768x1024]f32 -> bf16 WMMA -> 428 fused output cols ---
// One column panel spans ALL 448 columns: x is streamed from HBM exactly once.
__global__ __launch_bounds__(256)
void gemm_bf16(const float* __restrict__ x, const unsigned short* __restrict__ Wp,
               const float* __restrict__ bp, float* __restrict__ out) {
    __shared__ unsigned short As[BM][LDP];   // 10 KB bf16 activations
    __shared__ unsigned short Bs[BN][LDP];   // 35 KB bf16 weights (col-major by n)

    const int tid     = threadIdx.x;
    const int lane    = tid & 31;
    const int wave    = tid >> 5;
    const int waveRow = wave & 3;            // 4 row groups of 32 rows
    const int waveCol = wave >> 2;           // 2 col groups of 224 cols
    const int rowBase = blockIdx.x * BM;
    const int m       = lane & 15;
    const int h       = lane >> 4;

    v8f acc[2][NI] = {};

    for (int kt = 0; kt < DDIM; kt += BK) {
        // stage A tile: 128x32 f32 -> bf16, 4x float4 per thread
        #pragma unroll
        for (int i = 0; i < 4; ++i) {
            int idx = tid + i * 256;                 // 0..1023
            int r   = idx >> 3;                      // 0..127
            int kq  = (idx & 7) << 2;                // 0,4,...,28
            f32x4 f = *(const f32x4*)(x + (size_t)(rowBase + r) * DDIM + kt + kq);
            u32x2 pk;
            pk.x = (unsigned int)f2bf(f.x) | ((unsigned int)f2bf(f.y) << 16);
            pk.y = (unsigned int)f2bf(f.z) | ((unsigned int)f2bf(f.w) << 16);
            *(u32x2*)&As[r][kq] = pk;
        }
        // stage B tile: 448 cols x 32 k, 7x 16B loads per thread (L2 resident)
        #pragma unroll
        for (int i = 0; i < 7; ++i) {
            int c   = tid + i * 256;                 // 0..1791
            int col = c >> 2;
            int kq  = (c & 3) << 3;                  // 0,8,16,24
            *(u32x4*)&Bs[col][kq] =
                *(const u32x4*)(Wp + (size_t)col * DDIM + kt + kq);
        }
        // prefetch next A tile (global_prefetch_b8)
        if (kt + BK < DDIM) {
            const float* pf = x + (size_t)(rowBase + (tid >> 1)) * DDIM + kt + BK
                                + ((tid & 1) << 4);
            __builtin_prefetch(pf, 0, 0);
        }
        __syncthreads();

        // A fragments: lane m = row in tile; K chunk [h*16,+16)
        v16bf a[2];
        #pragma unroll
        for (int mi = 0; mi < 2; ++mi) {
            const unsigned short* p = &As[waveRow * 32 + mi * 16 + m][0];
            union { u32x4 q[2]; v16bf v; } U;
            U.q[0] = *(const u32x4*)(p + h * 16);
            U.q[1] = *(const u32x4*)(p + h * 16 + 8);
            a[mi] = U.v;
        }
        // 14 B fragments, 28 WMMAs. B is software-pipelined one step ahead so
        // each ds_load's latency is covered by the previous WMMA pair's issue.
        const unsigned short* bbase = &Bs[waveCol * 224 + m][h * 16];
        union { u32x4 q[2]; v16bf v; } B0, B1;
        B0.q[0] = *(const u32x4*)(bbase);
        B0.q[1] = *(const u32x4*)(bbase + 8);
        v16bf b_cur = B0.v;
        #pragma unroll
        for (int ni = 0; ni < NI; ++ni) {
            v16bf b_nxt = b_cur;
            if (ni + 1 < NI) {
                const unsigned short* p = bbase + (size_t)(ni + 1) * 16 * LDP;
                B1.q[0] = *(const u32x4*)(p);
                B1.q[1] = *(const u32x4*)(p + 8);
                b_nxt = B1.v;
            }
            acc[0][ni] = __builtin_amdgcn_wmma_f32_16x16x32_bf16(
                false, a[0], false, b_cur, (short)0, acc[0][ni], false, false);
            acc[1][ni] = __builtin_amdgcn_wmma_f32_16x16x32_bf16(
                false, a[1], false, b_cur, (short)0, acc[1][ni], false, false);
            b_cur = b_nxt;
        }
        __syncthreads();
    }

    // epilogue: +bias, remap GEMM col j -> output col, store
    #pragma unroll
    for (int ni = 0; ni < NI; ++ni) {
        int j = waveCol * 224 + ni * 16 + m;              // lane&15 = column
        if (j >= JTOT) continue;
        float bj = bp[j];
        int oc = (j < SNUM) ? j : j + (CNUM + 1);         // fine block (81) inserted
        #pragma unroll
        for (int mi = 0; mi < 2; ++mi) {
            #pragma unroll
            for (int v = 0; v < 8; ++v) {                 // VGPR v -> row v + h*8
                int r = rowBase + waveRow * 32 + mi * 16 + h * 8 + v;
                out[(size_t)r * OSTR + oc] = acc[mi][ni][v] + bj;
            }
        }
    }
}

// ---- 3) per-row: argmax(27) -> mask, 3-way softmax, permutation scatter -----
__global__ __launch_bounds__(256)
void row_epilogue(float* __restrict__ out, const int* __restrict__ cid) {
    int row  = blockIdx.x * 8 + (threadIdx.x >> 5);
    int lane = threadIdx.x & 31;
    float* o = out + (size_t)row * OSTR;

    float sv = (lane < SNUM) ? o[lane] : -__builtin_inff();
    int   si = lane;
    #pragma unroll
    for (int off = 16; off > 0; off >>= 1) {              // first-index tie-break
        float ov = __shfl_xor(sv, off, 32);
        int   oi = __shfl_xor(si, off, 32);
        if (ov > sv || (ov == sv && oi < si)) { sv = ov; si = oi; }
    }
    if (lane < SNUM) {
        int s = lane;
        float l0 = o[108 + 3 * s + 0];
        float l1 = o[108 + 3 * s + 1];
        float l2 = o[108 + 3 * s + 2];
        float mx = fmaxf(l0, fmaxf(l1, l2));
        float e0 = __expf(l0 - mx), e1 = __expf(l1 - mx), e2 = __expf(l2 - mx);
        float inv = 1.0f / (e0 + e1 + e2);
        float sc  = (s == si) ? 1.0f : 0.01f;
        o[SNUM + cid[3 * s + 0]] = fminf(e0 * inv * sc, 1.0f);
        o[SNUM + cid[3 * s + 1]] = fminf(e1 * inv * sc, 1.0f);
        o[SNUM + cid[3 * s + 2]] = fminf(e2 * inv * sc, 1.0f);
    }
}

extern "C" void kernel_launch(void* const* d_in, const int* in_sizes, int n_in,
                              void* d_out, int out_size, void* d_ws, size_t ws_size,
                              hipStream_t stream) {
    const float* x   = (const float*)d_in[0];
    const float* Wc  = (const float*)d_in[1];
    const float* bc  = (const float*)d_in[2];
    const float* Wf  = (const float*)d_in[3];
    const float* bfg = (const float*)d_in[4];
    const float* Wb  = (const float*)d_in[5];
    const float* bb  = (const float*)d_in[6];
    const int*   cid = (const int*)d_in[7];
    float* out = (float*)d_out;

    unsigned short* Wp = (unsigned short*)d_ws;                       // JPAD*DDIM bf16
    float*          bp = (float*)((char*)d_ws + (size_t)JPAD * DDIM * 2);

    pack_weights<<<(JPAD * DDIM) / 256, 256, 0, stream>>>(Wc, bc, Wf, bfg, Wb, bb, Wp, bp);
    gemm_bf16<<<NROW / BM, 256, 0, stream>>>(x, Wp, bp, out);
    row_epilogue<<<NROW / 8, 256, 0, stream>>>(out, cid);
}
